// DynMoE_53798760350315
// MI455X (gfx1250) — compile-verified, matrix operand order checked by reading, not run on hardware
//
#include <hip/hip_runtime.h>
#include <hip/hip_bf16.h>
#include <math.h>

// Problem dims (fixed by reference: B=2,T=2048,C=1024,E=8,I=2048)
#define N_TOK 4096
#define C_DIM 1024
#define I_DIM 2048
#define E_NUM 8

typedef __attribute__((ext_vector_type(16))) __bf16 v16bf;
typedef __attribute__((ext_vector_type(8)))  __bf16 bf16x8;
typedef __attribute__((ext_vector_type(8)))  float  v8f;
typedef __attribute__((ext_vector_type(4)))  unsigned int u32x4;
typedef __attribute__((ext_vector_type(8)))  int i32x8;
typedef __attribute__((ext_vector_type(4)))  int i32x4;

#if defined(__has_builtin)
#if __has_builtin(__builtin_amdgcn_tensor_load_to_lds) && __has_builtin(__builtin_amdgcn_s_wait_tensorcnt)
#define HAS_TDM 1
#endif
#endif
#ifndef HAS_TDM
#define HAS_TDM 0
#endif

union FragBF { v16bf v; bf16x8 h[2]; };

__device__ __forceinline__ unsigned short f2bf(float f) {
  unsigned u = __float_as_uint(f);
  u += 0x7FFFu + ((u >> 16) & 1u);          // round-to-nearest-even
  return (unsigned short)(u >> 16);
}

// Load one 16x32 bf16 WMMA A/B fragment from an LDS tile stored row-major
// with K-stride 32. Per ISA 7.12.2 (16-bit A 16x32): lanes 0-15 hold
// K[0..7],[16..23] of row=lane, lanes 16-31 hold K[8..15],[24..31] of
// row=lane-16 -> two ds_load_b128 per lane.
__device__ __forceinline__ v16bf load_frag(const unsigned short* base, int lane) {
  const unsigned short* p = base + (lane & 15) * 32 + ((lane >> 4) << 3);
  FragBF f;
  f.h[0] = *(const bf16x8*)(p);
  f.h[1] = *(const bf16x8*)(p + 16);
  return f.v;
}

// 12 ds_load_b128 + 8 v_wmma per call (2x4 16x16 tiles per wave, K=32).
__device__ __forceinline__ void compute8(const unsigned short* As, const unsigned short* Bs,
                                         v8f acc[2][4], int waveM, int waveN, int lane) {
  v16bf af[2], bfr[4];
#pragma unroll
  for (int i = 0; i < 2; i++)
    af[i] = load_frag(As + (waveM * 32 + i * 16) * 32, lane);
#pragma unroll
  for (int j = 0; j < 4; j++)
    bfr[j] = load_frag(Bs + (waveN * 64 + j * 16) * 32, lane);
#pragma unroll
  for (int i = 0; i < 2; i++)
#pragma unroll
    for (int j = 0; j < 4; j++)
      acc[i][j] = __builtin_amdgcn_wmma_f32_16x16x32_bf16(
          false, af[i], false, bfr[j], (short)0, acc[i][j], false, false);
}

#if HAS_TDM
// TDM: DMA a 2D tile [128 rows x 32 bf16], row stride K elems, packed into LDS
// at lds_off (row-major, 64B per row) -- exactly the fragment-friendly layout.
// D# fields per ISA 8.3/8.4: group0={count=1, lds_addr, global_addr, type=2},
// group1={data_size=2B, tensor_dim0=K, tensor_dim1=128, tile_dim0=32,
// tile_dim1=128, tensor_dim0_stride=K}. Groups 2/3 unused (2D tensor).
// 6-arg therock-10.0 builtin form:
//   (u32x4 g0, i32x8 g1, i32x4 g2, i32x4 g3, i32x8 pad, i32 cpol)
__device__ __forceinline__ void tdm_load_tile(unsigned lds_off, const void* gptr, int K) {
  unsigned long long ga = (unsigned long long)(uintptr_t)gptr;
  u32x4 g0;
  g0[0] = 1u;                                              // count=1, user mode
  g0[1] = lds_off;                                         // lds_addr
  g0[2] = (unsigned)ga;                                    // global_addr[31:0]
  g0[3] = ((unsigned)(ga >> 32) & 0x01FFFFFFu) | 0x80000000u; // addr[56:32] | type=2
  i32x8 g1;
  g1[0] = 0x10000;        // workgroup_mask=0, data_size=1 (2 bytes)
  g1[1] = K << 16;        // tensor_dim0[15:0]=K (atomic_barrier_addr=0)
  g1[2] = 128 << 16;      // tensor_dim0[31:16]=0 | tensor_dim1[15:0]=128
  g1[3] = 32 << 16;       // tensor_dim1[31:16]=0 | tile_dim0=32
  g1[4] = 128;            // tile_dim1=128, tile_dim2=0
  g1[5] = K;              // tensor_dim0_stride[31:0]=K
  g1[6] = 0;              // stride0 hi | stride1 lo
  g1[7] = 0;              // stride1 hi
  i32x4 z4 = {0, 0, 0, 0};
  i32x8 z8 = {0, 0, 0, 0, 0, 0, 0, 0};
  __builtin_amdgcn_tensor_load_to_lds(g0, g1, z4, z4, z8, 0);
}
#endif

// K-loop producing acc[2][4]. lds points at 4 x 4096 ushort (32KB):
// TDM path double-buffers (A0,B0 | A1,B1); fallback uses first 16KB.
// All LDS offsets are compile-time constants so clang keeps AS(3) provenance
// and emits ds_load_b128 (not flat_load) for the fragment reads.
__device__ __forceinline__ void mma_loop(const unsigned short* __restrict__ Ag,
                                         const unsigned short* __restrict__ Bg,
                                         int K, unsigned short* lds,
                                         v8f acc[2][4], int t) {
  const int lane = t & 31;
  const int w = t >> 5;
  const int waveM = w & 3;      // 4 waves along M (32 rows each)
  const int waveN = w >> 2;     // 2 waves along N (64 cols each)

#if HAS_TDM
  const unsigned ldsA0 = (unsigned)(uintptr_t)(lds);
  const unsigned ldsB0 = (unsigned)(uintptr_t)(lds + 4096);
  const unsigned ldsA1 = (unsigned)(uintptr_t)(lds + 2 * 4096);
  const unsigned ldsB1 = (unsigned)(uintptr_t)(lds + 3 * 4096);
  const int nk = K / 32;              // 32 or 64 here -> always even
  if (w == 0) {                       // one wave drives the DMA engine
    tdm_load_tile(ldsA0, Ag, K);
    tdm_load_tile(ldsB0, Bg, K);
  }
  for (int s = 0; s < nk; s += 2) {
    // ---- stage s: compute from buffer 0, DMA stage s+1 into buffer 1
    if (w == 0) {
      tdm_load_tile(ldsA1, Ag + (size_t)(s + 1) * 32, K);
      tdm_load_tile(ldsB1, Bg + (size_t)(s + 1) * 32, K);
      __builtin_amdgcn_s_wait_tensorcnt(2);   // stage s tiles landed
    }
    __syncthreads();
    compute8(lds, lds + 4096, acc, waveM, waveN, lane);
    __syncthreads();                          // buffer 0 free for next DMA
    // ---- stage s+1: compute from buffer 1, DMA stage s+2 into buffer 0
    if (w == 0) {
      if (s + 2 < nk) {
        tdm_load_tile(ldsA0, Ag + (size_t)(s + 2) * 32, K);
        tdm_load_tile(ldsB0, Bg + (size_t)(s + 2) * 32, K);
        __builtin_amdgcn_s_wait_tensorcnt(2); // stage s+1 tiles landed
      } else {
        __builtin_amdgcn_s_wait_tensorcnt(0);
      }
    }
    __syncthreads();
    compute8(lds + 2 * 4096, lds + 3 * 4096, acc, waveM, waveN, lane);
    __syncthreads();                          // buffer 1 free for next DMA
  }
#else
  unsigned short* As = lds;
  unsigned short* Bs = lds + 4096;
  const int id0 = t, id1 = t + 256;   // 512 x 16B chunks per 8KB tile
  const size_t off0 = (size_t)(id0 >> 2) * K + (size_t)(id0 & 3) * 8;
  const size_t off1 = (size_t)(id1 >> 2) * K + (size_t)(id1 & 3) * 8;
  uint4 ra0 = *(const uint4*)(Ag + off0);
  uint4 ra1 = *(const uint4*)(Ag + off1);
  uint4 rb0 = *(const uint4*)(Bg + off0);
  uint4 rb1 = *(const uint4*)(Bg + off1);
  for (int k0 = 0; k0 < K; k0 += 32) {
    __syncthreads();
    ((uint4*)As)[id0] = ra0; ((uint4*)As)[id1] = ra1;
    ((uint4*)Bs)[id0] = rb0; ((uint4*)Bs)[id1] = rb1;
    __syncthreads();
    int kn = k0 + 32;
    if (kn < K) {                     // register double-buffer next tile
      ra0 = *(const uint4*)(Ag + off0 + kn);
      ra1 = *(const uint4*)(Ag + off1 + kn);
      rb0 = *(const uint4*)(Bg + off0 + kn);
      rb1 = *(const uint4*)(Bg + off1 + kn);
      if (kn + 32 < K) {
        __builtin_prefetch(Ag + off0 + kn + 32, 0, 1);
        __builtin_prefetch(Bg + off0 + kn + 32, 0, 1);
      }
    }
    compute8(As, Bs, acc, waveM, waveN, lane);
  }
#endif
}

// ---------------------------------------------------------------- conversions
__global__ __launch_bounds__(256) void cvt_bf16_kernel(
    const float4* __restrict__ in, ushort4* __restrict__ out, int n4) {
  int i = blockIdx.x * blockDim.x + threadIdx.x;
  if (i < n4) {
    float4 v = in[i];
    ushort4 o;
    o.x = f2bf(v.x); o.y = f2bf(v.y); o.z = f2bf(v.z); o.w = f2bf(v.w);
    out[i] = o;
  }
}

// in: [rows][cols] f32, out: [cols][rows] bf16 (per blockIdx.z matrix)
__global__ __launch_bounds__(256) void transpose_cvt_kernel(
    const float* __restrict__ in, unsigned short* __restrict__ out,
    int rows, int cols) {
  __shared__ float tile[32][33];
  size_t mat = (size_t)blockIdx.z * rows * cols;
  in  += mat;
  out += mat;
  int c0 = blockIdx.x * 32, r0 = blockIdx.y * 32;
#pragma unroll
  for (int j = 0; j < 4; j++) {
    int r = r0 + threadIdx.y + j * 8;
    tile[threadIdx.y + j * 8][threadIdx.x] = in[(size_t)r * cols + c0 + threadIdx.x];
  }
  __syncthreads();
#pragma unroll
  for (int j = 0; j < 4; j++) {
    int c = c0 + threadIdx.y + j * 8;
    out[(size_t)c * rows + r0 + threadIdx.x] = f2bf(tile[threadIdx.x][threadIdx.y + j * 8]);
  }
}

// ---------------------------------------------------------------- gating
// One wave (32 lanes) per token: cosine logits vs 8 normalized sim columns,
// hard threshold, top-min_experts fallback. fp32 throughout (tiny work).
__global__ __launch_bounds__(256) void gating_kernel(
    const float* __restrict__ H, const float* __restrict__ S,
    const float* __restrict__ gates, const float* __restrict__ temp,
    const float* __restrict__ mask, const int* __restrict__ minExp,
    float* __restrict__ logits_out, float* __restrict__ act_out,
    float* __restrict__ scale_out) {
  const int lane = threadIdx.x & 31;
  const int wid  = threadIdx.x >> 5;
  const int n    = blockIdx.x * 8 + wid;
  const float* h = H + (size_t)n * C_DIM;

  float hh = 0.f, dot[8], ss[8];
#pragma unroll
  for (int e = 0; e < 8; e++) { dot[e] = 0.f; ss[e] = 0.f; }

  for (int c = lane; c < C_DIM; c += 32) {
    float x = h[c];
    hh += x * x;
    float4 s0 = *(const float4*)(S + (size_t)c * 8);
    float4 s1 = *(const float4*)(S + (size_t)c * 8 + 4);
    float sv[8] = {s0.x, s0.y, s0.z, s0.w, s1.x, s1.y, s1.z, s1.w};
#pragma unroll
    for (int e = 0; e < 8; e++) { dot[e] += x * sv[e]; ss[e] += sv[e] * sv[e]; }
  }
#pragma unroll
  for (int off = 16; off > 0; off >>= 1) {
    hh += __shfl_xor(hh, off, 32);
#pragma unroll
    for (int e = 0; e < 8; e++) {
      dot[e] += __shfl_xor(dot[e], off, 32);
      ss[e]  += __shfl_xor(ss[e], off, 32);
    }
  }
  if (lane == 0) {
    const float EPSN = 1e-12f;
    float ls = 1.0f / (1.0f + __expf(-temp[0]));   // sigmoid(temperature)
    float hn = fmaxf(sqrtf(hh), EPSN);
    float lg[8], act[8];
    float sum = 0.f;
#pragma unroll
    for (int e = 0; e < 8; e++) {
      float sn = fmaxf(sqrtf(ss[e]), EPSN);
      float l  = (dot[e] / (hn * sn)) * mask[e];
      lg[e] = l;
      float gated = fmaxf(l - gates[e] * ls, 0.f);
      act[e] = gated > 0.f ? 1.f : 0.f;
      sum += act[e];
    }
    if (sum == 0.f) {  // fallback: top-min_experts by logits
      int k = minExp[0];
      bool chosen[8] = {false, false, false, false, false, false, false, false};
      for (int kk = 0; kk < k; kk++) {
        int best = -1; float bv = -1e30f;
        for (int e = 0; e < 8; e++)
          if (!chosen[e] && lg[e] > bv) { bv = lg[e]; best = e; }
        if (best >= 0) { chosen[best] = true; act[best] = 1.f; sum += 1.f; }
      }
    }
    float inv = 1.0f / fmaxf(sum, 1.0f);
#pragma unroll
    for (int e = 0; e < 8; e++) {
      logits_out[(size_t)n * 8 + e] = lg[e];
      act_out  [(size_t)n * 8 + e] = act[e];
      scale_out[(size_t)n * 8 + e] = act[e] * inv;   // folded act/num_act
    }
  }
}

// ---------------------------------------------------------------- GEMM 1
// inter = gelu(Hb[M,K] @ W1t^T), Bt given as [Nc,K] (pre-transposed), out bf16.
__global__ __launch_bounds__(256) void gemm1_gelu_kernel(
    const unsigned short* __restrict__ A,   // [M x K] bf16
    const unsigned short* __restrict__ Bt,  // [Nc x K] bf16
    unsigned short* __restrict__ Out,       // [M x ldo] bf16
    int K, int ldo) {
  __shared__ __align__(16) unsigned short lds[4 * 4096];   // 32 KB
  const int t = threadIdx.x;
  const int lane = t & 31;
  const int w = t >> 5;
  const int waveM = w & 3, waveN = w >> 2;
  const size_t bm = (size_t)blockIdx.y * 128;
  const size_t bn = (size_t)blockIdx.x * 128;

  v8f acc[2][4] = {};
  mma_loop(A + bm * K, Bt + bn * K, K, lds, acc, t);

  // C/D layout: VGPR v holds M=v (lanes 0-15) / M=v+8 (lanes 16-31), N=lane%16
  const int hi = (lane >> 4) << 3;
  const int ln = lane & 15;
#pragma unroll
  for (int i = 0; i < 2; i++)
#pragma unroll
    for (int j = 0; j < 4; j++)
#pragma unroll
      for (int v = 0; v < 8; v++) {
        int row = waveM * 32 + i * 16 + v + hi;
        int col = waveN * 64 + j * 16 + ln;
        float x = acc[i][j][v];
        float g = 0.5f * x * (1.0f + erff(x * 0.70710678118654752f));  // exact gelu
        Out[(bm + row) * (size_t)ldo + bn + col] = f2bf(g);
      }
}

// ---------------------------------------------------------------- GEMM 2
// final += (act/num_act)[token,e] * (inter[M,K] @ W2t^T); f32 output.
__global__ __launch_bounds__(256) void gemm2_acc_kernel(
    const unsigned short* __restrict__ A,   // inter [M x K] bf16
    const unsigned short* __restrict__ Bt,  // W2t [Nc x K] bf16
    float* __restrict__ Out,                // final [M x ldo] f32
    const float* __restrict__ scale,        // [M x E]
    int K, int ldo, int e, int accum) {
  __shared__ __align__(16) unsigned short lds[4 * 4096];   // 32 KB
  __shared__ float sS[128];
  const int t = threadIdx.x;
  const int lane = t & 31;
  const int w = t >> 5;
  const int waveM = w & 3, waveN = w >> 2;
  const size_t bm = (size_t)blockIdx.y * 128;
  const size_t bn = (size_t)blockIdx.x * 128;

  if (t < 128) sS[t] = scale[(bm + t) * E_NUM + e];

  v8f acc[2][4] = {};
  mma_loop(A + bm * K, Bt + bn * K, K, lds, acc, t);

  const int hi = (lane >> 4) << 3;
  const int ln = lane & 15;
#pragma unroll
  for (int i = 0; i < 2; i++)
#pragma unroll
    for (int j = 0; j < 4; j++)
#pragma unroll
      for (int v = 0; v < 8; v++) {
        int row = waveM * 32 + i * 16 + v + hi;
        int col = waveN * 64 + j * 16 + ln;
        float val = acc[i][j][v] * sS[row];
        size_t o = (bm + row) * (size_t)ldo + bn + col;
        if (accum) Out[o] += val; else Out[o] = val;
      }
}

// ---------------------------------------------------------------- launch
extern "C" void kernel_launch(void* const* d_in, const int* in_sizes, int n_in,
                              void* d_out, int out_size, void* d_ws, size_t ws_size,
                              hipStream_t stream) {
  const float* H     = (const float*)d_in[0];
  const float* S     = (const float*)d_in[1];
  const float* gates = (const float*)d_in[2];
  const float* temp  = (const float*)d_in[3];
  const float* mask  = (const float*)d_in[4];
  const float* w1    = (const float*)d_in[5];
  const float* w2    = (const float*)d_in[6];
  const int*   minE  = (const int*)d_in[7];

  float* finalOut  = (float*)d_out;                         // [N_TOK x C]
  float* logitsOut = finalOut + (size_t)N_TOK * C_DIM;      // [N_TOK x E]
  float* actOut    = logitsOut + (size_t)N_TOK * E_NUM;     // [N_TOK x E]

  char* ws = (char*)d_ws;
  size_t off = 0;
  auto alloc = [&](size_t bytes) {
    void* p = ws + off;
    off = (off + bytes + 255) & ~(size_t)255;
    return p;
  };
  unsigned short* Hb  = (unsigned short*)alloc((size_t)N_TOK * C_DIM * 2);
  unsigned short* W1t = (unsigned short*)alloc((size_t)E_NUM * I_DIM * C_DIM * 2); // [E][I][C]
  unsigned short* W2t = (unsigned short*)alloc((size_t)E_NUM * C_DIM * I_DIM * 2); // [E][C][I]
  unsigned short* Ib  = (unsigned short*)alloc((size_t)N_TOK * I_DIM * 2);         // reused per expert
  float* scale        = (float*)alloc((size_t)N_TOK * E_NUM * 4);

  int n4 = N_TOK * C_DIM / 4;
  cvt_bf16_kernel<<<(n4 + 255) / 256, 256, 0, stream>>>((const float4*)H, (ushort4*)Hb, n4);
  // w1 [E][C][I] -> W1t [E][I][C];  w2 [E][I][C] -> W2t [E][C][I]
  transpose_cvt_kernel<<<dim3(I_DIM / 32, C_DIM / 32, E_NUM), dim3(32, 8), 0, stream>>>(w1, W1t, C_DIM, I_DIM);
  transpose_cvt_kernel<<<dim3(C_DIM / 32, I_DIM / 32, E_NUM), dim3(32, 8), 0, stream>>>(w2, W2t, I_DIM, C_DIM);
  gating_kernel<<<N_TOK / 8, 256, 0, stream>>>(H, S, gates, temp, mask, minE,
                                               logitsOut, actOut, scale);

  for (int e = 0; e < E_NUM; e++) {
    gemm1_gelu_kernel<<<dim3(I_DIM / 128, N_TOK / 128), 256, 0, stream>>>(
        Hb, W1t + (size_t)e * I_DIM * C_DIM, Ib, C_DIM, I_DIM);
    gemm2_acc_kernel<<<dim3(C_DIM / 128, N_TOK / 128), 256, 0, stream>>>(
        Ib, W2t + (size_t)e * C_DIM * I_DIM, finalOut, scale, I_DIM, C_DIM, e, e != 0);
  }
}
